// Transducer_962072674643
// MI455X (gfx1250) — matrix operand order repaired and checked by reading, not compiled, not run on hardware
//
#include <hip/hip_runtime.h>
#include <cstdint>
#include <cstddef>

// ---------------------------------------------------------------------------
// Types / WMMA helpers (CDNA5 gfx1250, wave32)
// ---------------------------------------------------------------------------
typedef __bf16 bf16_t;
typedef __attribute__((ext_vector_type(16))) __bf16        bf16x16;
typedef __attribute__((ext_vector_type(8)))  float         f32x8;
typedef __attribute__((ext_vector_type(4)))  unsigned int  u32x4;

#define DEVFN static __device__ __forceinline__

// Hardware tanh (V_TANH_F32) when the builtin exists; libm fallback otherwise.
#if defined(__has_builtin)
#  if __has_builtin(__builtin_amdgcn_tanhf)
#    define FAST_TANH(x) __builtin_amdgcn_tanhf(x)
#  elif __has_builtin(__builtin_amdgcn_tanh_f32)
#    define FAST_TANH(x) __builtin_amdgcn_tanh_f32(x)
#  endif
#endif
#ifndef FAST_TANH
#  define FAST_TANH(x) tanhf(x)
#endif

DEVFN f32x8 wmma_bf16(bf16x16 a, bf16x16 b, f32x8 c) {
  // (neg_a, A, neg_b, B, c_mod, C, reuse_a, reuse_b)
  return __builtin_amdgcn_wmma_f32_16x16x32_bf16(false, a, false, b, (short)0, c,
                                                 false, false);
}

union FragU { bf16x16 v; u32x4 q[2]; };

// A-matrix fragment, 16x32 bf16 (M x K), per ISA 7.12.2 layout:
// lane m = lane&15, half = lane>>4; VGPR0..3 hold K = half*8 + [0..8),
// VGPR4..7 hold K = 16 + half*8 + [0..8)  -> two contiguous 16B loads.
DEVFN bf16x16 load_a_g(const bf16_t* __restrict__ A, int lda, int k0, int lane) {
  const int m = lane & 15, half = lane >> 4;
  const bf16_t* p = A + (size_t)m * lda + (k0 + half * 8);
  FragU u;
  u.q[0] = *(const u32x4*)(p);
  u.q[1] = *(const u32x4*)(p + 16);
  return u.v;
}

// B-matrix fragment, 32x16 bf16 (K x N), computing A @ W^T with W[N][K]:
// lane n = lane&15, half = lane>>4; VGPR0..3 hold K = half*8 + [0..8),
// VGPR4..7 hold K = 16 + half*8 + [0..8) -> two contiguous 16B loads.
DEVFN bf16x16 load_b_g(const bf16_t* __restrict__ W, int ldw, int k0, int lane) {
  const int n = lane & 15, kh = k0 + ((lane >> 4) << 3);
  const bf16_t* p = W + (size_t)n * ldw + kh;
  FragU u;
  u.q[0] = *(const u32x4*)(p);
  u.q[1] = *(const u32x4*)(p + 16);
  return u.v;
}

DEVFN float sigf(float x) { return 1.0f / (1.0f + __expf(-x)); }

// Generation-counter grid barrier (16 persistent WGs).
DEVFN void grid_barrier(unsigned* cnt, unsigned* gen, unsigned nwg) {
  __syncthreads();
  if (threadIdx.x == 0) {
    __threadfence();
    unsigned g   = atomicAdd(gen, 0u);
    unsigned old = atomicAdd(cnt, 1u);
    if (old == nwg - 1u) {
      atomicExch(cnt, 0u);
      __threadfence();
      atomicAdd(gen, 1u);
    } else {
      while (atomicAdd(gen, 0u) == g) { __builtin_amdgcn_s_sleep(2); }
    }
    __threadfence();
  }
  __syncthreads();
}

// ---------------------------------------------------------------------------
// Elementwise kernels
// ---------------------------------------------------------------------------
// dst[R,Cp] bf16 = src[r, src_off + c] (c<C), zero-padded to Cp.
__global__ __launch_bounds__(256) void k_convert(const float* __restrict__ src,
                                                 bf16_t* __restrict__ dst,
                                                 int R, int C, int src_ld,
                                                 int src_off, int Cp) {
  long long idx = (long long)blockIdx.x * blockDim.x + threadIdx.x;
  long long total = (long long)R * Cp;
  if (idx >= total) return;
  int r = (int)(idx / Cp);
  int c = (int)(idx % Cp);
  float v = (c < C) ? src[(size_t)r * src_ld + src_off + c] : 0.0f;
  dst[idx] = (bf16_t)v;
}

// Decoder input: e[528,512] bf16; row r = b*65+u; token = (u==0)?BOS:ys[b][u-1].
__global__ __launch_bounds__(256) void k_embed(const int* __restrict__ ys,
                                               const float* __restrict__ embed,
                                               bf16_t* __restrict__ e) {
  int idx = blockIdx.x * 256 + threadIdx.x;  // over 528*512
  if (idx >= 528 * 512) return;
  int r = idx >> 9, c = idx & 511;
  if (r >= 520) { e[idx] = (bf16_t)0.0f; return; }
  int b = r / 65, u = r % 65;
  int tok = (u == 0) ? 0 : ys[b * 64 + (u - 1)];
  e[idx] = (bf16_t)embed[(size_t)tok * 512 + c];
}

// ---------------------------------------------------------------------------
// Generic GEMM: C[M,N](f32 or bf16) = A_bf16[M,K] @ W_bf16[N,K]^T + bias
// Block = 4 waves, each wave computes a 16x32 output strip (2 N-tiles).
// grid = (M/16, N/128). M multiple of 16, K multiple of 32, N multiple of 128.
// ---------------------------------------------------------------------------
template <bool OUT_BF16>
__global__ __launch_bounds__(128) void k_gemm(const bf16_t* __restrict__ A,
                                              const bf16_t* __restrict__ W,
                                              const float* __restrict__ bias,
                                              void* __restrict__ Cptr,
                                              int K, int N) {
  const int tid = threadIdx.x;
  const int wave = tid >> 5, lane = tid & 31;
  const int m0 = blockIdx.x * 16;
  const int n0 = blockIdx.y * 128 + wave * 32;
  const bf16_t* Ar = A + (size_t)m0 * K;
  const bf16_t* W0 = W + (size_t)n0 * K;
  const bf16_t* W1 = W + (size_t)(n0 + 16) * K;
  // Pull the streamed weight rows toward the caches (global_prefetch_b8).
  __builtin_prefetch(W0 + (size_t)(lane & 15) * K, 0, 1);
  __builtin_prefetch(W1 + (size_t)(lane & 15) * K, 0, 1);
  f32x8 acc0 = {}, acc1 = {};
  for (int k0 = 0; k0 < K; k0 += 32) {
    bf16x16 a  = load_a_g(Ar, K, k0, lane);
    bf16x16 b0 = load_b_g(W0, K, k0, lane);
    bf16x16 b1 = load_b_g(W1, K, k0, lane);
    acc0 = wmma_bf16(a, b0, acc0);
    acc1 = wmma_bf16(a, b1, acc1);
  }
  const int n = lane & 15, mb = (lane >> 4) * 8;
  const float bs0 = bias ? bias[n0 + n] : 0.0f;
  const float bs1 = bias ? bias[n0 + 16 + n] : 0.0f;
#pragma unroll
  for (int r = 0; r < 8; ++r) {
    size_t row = (size_t)(m0 + mb + r);
    float v0 = acc0[r] + bs0;
    float v1 = acc1[r] + bs1;
    if (OUT_BF16) {
      bf16_t* Cb = (bf16_t*)Cptr;
      Cb[row * N + n0 + n]      = (bf16_t)v0;
      Cb[row * N + n0 + 16 + n] = (bf16_t)v1;
    } else {
      float* Cf = (float*)Cptr;
      Cf[row * N + n0 + n]      = v0;
      Cf[row * N + n0 + 16 + n] = v1;
    }
  }
}

// ---------------------------------------------------------------------------
// Persistent cooperative LSTM scan.
// 16 WGs x 256 threads (8 waves). WG w owns hidden units [w*32, w*32+32).
// Per step: gates_pre = h @ Whh^T for this WG's 128 gate columns
// (wave v -> gate g=v>>1, 16-col block nblk=v&1) via WMMA (batch 8 padded
// to M=16 with zero rows), gate exchange via LDS, f32 cell update,
// bf16 h-state ping-pong in global, then grid barrier.
// xp: precomputed input projections + bias, f32 [B*S, 2048], row = b*S+t.
// ---------------------------------------------------------------------------
__global__ __launch_bounds__(256) void k_lstm_scan(const float* __restrict__ xp,
                                                   const bf16_t* __restrict__ Whh,
                                                   bf16_t* __restrict__ hseq,
                                                   bf16_t* hstate,  // [2][16][512]
                                                   float*  cstate,  // [8][512]
                                                   unsigned* cnt, unsigned* gen,
                                                   int S) {
  const int H = 512;
  const int wg = blockIdx.x;  // 0..15
  const int tid = threadIdx.x;
  const int wave = tid >> 5, lane = tid & 31;
  const int g = wave >> 1, nblk = wave & 1;
  const int n0 = wg * 32;
  const bf16_t* Wr = Whh + (size_t)(g * H + n0 + nblk * 16) * H;
  __shared__ float gl[4][32][8];  // [gate][unit in WG][batch]
  const int uu = tid >> 3, bb = tid & 7;  // consumer mapping: 32 units x 8 batch
  for (int t = 0; t < S; ++t) {
    const bf16_t* hread = hstate + (size_t)(t & 1) * 16 * H;
    bf16_t* hwrite      = hstate + (size_t)((t + 1) & 1) * 16 * H;
    f32x8 acc = {};
#pragma unroll 4
    for (int kt = 0; kt < 16; ++kt) {
      bf16x16 a = load_a_g(hread, H, kt * 32, lane);
      bf16x16 b = load_b_g(Wr, H, kt * 32, lane);
      acc = wmma_bf16(a, b, acc);
    }
    if (lane < 16) {  // valid batch rows (M=0..7) live in lanes 0..15
      int gc = nblk * 16 + lane;
#pragma unroll
      for (int r = 0; r < 8; ++r) gl[g][gc][r] = acc[r];
    }
    __syncthreads();
    {
      size_t xrow = (size_t)(bb * S + t) * 2048;
      int col = n0 + uu;
      float xi = xp[xrow + col];
      float xf = xp[xrow + H + col];
      float xg = xp[xrow + 2 * H + col];
      float xo = xp[xrow + 3 * H + col];
      float iv = sigf(gl[0][uu][bb] + xi);
      float fv = sigf(gl[1][uu][bb] + xf);
      float gv = FAST_TANH(gl[2][uu][bb] + xg);
      float ov = sigf(gl[3][uu][bb] + xo);
      float c = fv * cstate[bb * H + col] + iv * gv;
      cstate[bb * H + col] = c;
      float h = ov * FAST_TANH(c);
      bf16_t hb = (bf16_t)h;
      hwrite[(size_t)bb * H + col]         = hb;
      hseq[(size_t)(bb * S + t) * H + col] = hb;
    }
    grid_barrier(cnt, gen, gridDim.x);
  }
}

// ---------------------------------------------------------------------------
// Joint: logits[b,t,u,:] = tanh(pre_enc[b,t]+pre_dec[b,u]+jb1) @ jW2^T + jb2.
// Row index R = b*T*65 + t*65 + u matches the output layout exactly.
// WG = 256 threads (8 waves): build a 32x512 bf16 z-tile in LDS (v_tanh),
// then each wave computes 2 M-tiles x 2 N-tiles, sharing every B fragment
// across both M-tiles. grid = (133120/32, 1024/256).
// ---------------------------------------------------------------------------
__global__ __launch_bounds__(256) void k_joint(const float* __restrict__ preE,
                                               const float* __restrict__ preD,
                                               const float* __restrict__ jb1,
                                               const bf16_t* __restrict__ jW2b,
                                               const float* __restrict__ jb2,
                                               float* __restrict__ out) {
  const int J = 512, V = 1024, U1 = 65, T = 256;
  const int r0 = blockIdx.x * 32;          // global output row of this tile
  const int b  = r0 / (T * U1);
  const int rb = r0 % (T * U1);            // tiles never straddle b (16640%32==0)
  const int tid = threadIdx.x, wave = tid >> 5, lane = tid & 31;
  __shared__ bf16_t zt[32][512];           // 32 KB
  {
    int row = tid >> 3;                    // 0..31
    int c0  = (tid & 7) * 64;              // 0..448
    int rr  = rb + row;
    int t = rr / U1, u = rr % U1;
    const float4* pe = (const float4*)(preE + (size_t)(b * T + t) * J + c0);
    const float4* pd = (const float4*)(preD + (size_t)(b * U1 + u) * J + c0);
    const float4* pb = (const float4*)(jb1 + c0);
#pragma unroll 4
    for (int c4 = 0; c4 < 16; ++c4) {
      float4 e4 = pe[c4], d4 = pd[c4], b4 = pb[c4];
      int c = c0 + c4 * 4;
      zt[row][c + 0] = (bf16_t)FAST_TANH(e4.x + d4.x + b4.x);
      zt[row][c + 1] = (bf16_t)FAST_TANH(e4.y + d4.y + b4.y);
      zt[row][c + 2] = (bf16_t)FAST_TANH(e4.z + d4.z + b4.z);
      zt[row][c + 3] = (bf16_t)FAST_TANH(e4.w + d4.w + b4.w);
    }
  }
  __syncthreads();
  const int n0 = blockIdx.y * 256 + wave * 32;
  const bf16_t* W0 = jW2b + (size_t)n0 * J;
  const bf16_t* W1 = jW2b + (size_t)(n0 + 16) * J;
  const int m = lane & 15, half = lane >> 4;
  f32x8 acc00 = {}, acc01 = {}, acc10 = {}, acc11 = {};
  for (int kt = 0; kt < 16; ++kt) {
    int k0 = kt * 32;
    FragU a0, a1;  // A fragments gathered from LDS (ds_load_b128 x2 each)
    a0.q[0] = *(const u32x4*)(&zt[m][k0 + half * 8]);
    a0.q[1] = *(const u32x4*)(&zt[m][k0 + 16 + half * 8]);
    a1.q[0] = *(const u32x4*)(&zt[16 + m][k0 + half * 8]);
    a1.q[1] = *(const u32x4*)(&zt[16 + m][k0 + 16 + half * 8]);
    bf16x16 b0 = load_b_g(W0, J, k0, lane);
    bf16x16 b1 = load_b_g(W1, J, k0, lane);
    acc00 = wmma_bf16(a0.v, b0, acc00);
    acc01 = wmma_bf16(a0.v, b1, acc01);
    acc10 = wmma_bf16(a1.v, b0, acc10);
    acc11 = wmma_bf16(a1.v, b1, acc11);
  }
  const int n = lane & 15, mb = half * 8;
  const float bs0 = jb2[n0 + n], bs1 = jb2[n0 + 16 + n];
#pragma unroll
  for (int r = 0; r < 8; ++r) {
    size_t row0 = (size_t)(r0 + mb + r);
    size_t row1 = (size_t)(r0 + 16 + mb + r);
    out[row0 * V + n0 + n]      = acc00[r] + bs0;
    out[row0 * V + n0 + 16 + n] = acc01[r] + bs1;
    out[row1 * V + n0 + n]      = acc10[r] + bs0;
    out[row1 * V + n0 + 16 + n] = acc11[r] + bs1;
  }
}

// ---------------------------------------------------------------------------
// Host launcher
// ---------------------------------------------------------------------------
extern "C" void kernel_launch(void* const* d_in, const int* in_sizes, int n_in,
                              void* d_out, int out_size, void* d_ws, size_t ws_size,
                              hipStream_t stream) {
  (void)in_sizes; (void)n_in; (void)out_size; (void)ws_size;
  const float* xs     = (const float*)d_in[0];
  const int*   ys     = (const int*)  d_in[1];
  const float* eWih0  = (const float*)d_in[2];
  const float* eWhh0  = (const float*)d_in[3];
  const float* eb0    = (const float*)d_in[4];
  const float* eWih1  = (const float*)d_in[5];
  const float* eWhh1  = (const float*)d_in[6];
  const float* eb1    = (const float*)d_in[7];
  const float* eProjW = (const float*)d_in[8];
  const float* eProjb = (const float*)d_in[9];
  const float* embed  = (const float*)d_in[10];
  const float* dWih   = (const float*)d_in[11];
  const float* dWhh   = (const float*)d_in[12];
  const float* db     = (const float*)d_in[13];
  const float* dProjW = (const float*)d_in[14];
  const float* dProjb = (const float*)d_in[15];
  const float* jW1    = (const float*)d_in[16];
  const float* jb1    = (const float*)d_in[17];
  const float* jW2    = (const float*)d_in[18];
  const float* jb2    = (const float*)d_in[19];
  float* out = (float*)d_out;

  char* wsp = (char*)d_ws;
  auto alloc = [&](size_t bytes) -> void* {
    void* p = (void*)wsp;
    wsp += (bytes + 255) & ~((size_t)255);
    return p;
  };

  // bf16 weights
  bf16_t* wWih0 = (bf16_t*)alloc((size_t)2048 * 96 * 2);   // K padded 80->96
  bf16_t* wWhh0 = (bf16_t*)alloc((size_t)2048 * 512 * 2);
  bf16_t* wWih1 = (bf16_t*)alloc((size_t)2048 * 512 * 2);
  bf16_t* wWhh1 = (bf16_t*)alloc((size_t)2048 * 512 * 2);
  bf16_t* wdWih = (bf16_t*)alloc((size_t)2048 * 512 * 2);
  bf16_t* wdWhh = (bf16_t*)alloc((size_t)2048 * 512 * 2);
  bf16_t* wEPrj = (bf16_t*)alloc((size_t)512 * 512 * 2);
  bf16_t* wDPrj = (bf16_t*)alloc((size_t)512 * 512 * 2);
  bf16_t* wJ1e  = (bf16_t*)alloc((size_t)512 * 512 * 2);   // jW1[:, :512]
  bf16_t* wJ1d  = (bf16_t*)alloc((size_t)512 * 512 * 2);   // jW1[:, 512:]
  bf16_t* wJ2   = (bf16_t*)alloc((size_t)1024 * 512 * 2);
  // activations
  bf16_t* xsPad = (bf16_t*)alloc((size_t)2048 * 96 * 2);
  float*  xp    = (float*) alloc((size_t)2048 * 2048 * 4); // reused 3x
  bf16_t* h0seq = (bf16_t*)alloc((size_t)2048 * 512 * 2);
  bf16_t* h1seq = (bf16_t*)alloc((size_t)2048 * 512 * 2);
  bf16_t* eDec  = (bf16_t*)alloc((size_t)528 * 512 * 2);
  bf16_t* hdseq = (bf16_t*)alloc((size_t)528 * 512 * 2);
  bf16_t* hEnc  = (bf16_t*)alloc((size_t)2048 * 512 * 2);
  bf16_t* hDec  = (bf16_t*)alloc((size_t)528 * 512 * 2);
  float*  preE  = (float*) alloc((size_t)2048 * 512 * 4);
  float*  preD  = (float*) alloc((size_t)528 * 512 * 4);
  // scan state block: hstate[2][16][512]bf16 | cstate[8][512]f32 | counters
  char* stateBlk = (char*)alloc(49408);
  bf16_t*   hstate = (bf16_t*)stateBlk;
  float*    cstate = (float*)(stateBlk + 32768);
  unsigned* cnt    = (unsigned*)(stateBlk + 49152);
  unsigned* gen    = cnt + 1;

  auto cvt = [&](const float* src, bf16_t* dst, int R, int C, int ld, int off, int Cp) {
    long long total = (long long)R * Cp;
    int blocks = (int)((total + 255) / 256);
    k_convert<<<blocks, 256, 0, stream>>>(src, dst, R, C, ld, off, Cp);
  };

  // Weight / input conversions to bf16
  cvt(xs,     xsPad, 2048, 80, 80, 0, 96);
  cvt(eWih0,  wWih0, 2048, 80, 80, 0, 96);
  cvt(eWhh0,  wWhh0, 2048, 512, 512, 0, 512);
  cvt(eWih1,  wWih1, 2048, 512, 512, 0, 512);
  cvt(eWhh1,  wWhh1, 2048, 512, 512, 0, 512);
  cvt(dWih,   wdWih, 2048, 512, 512, 0, 512);
  cvt(dWhh,   wdWhh, 2048, 512, 512, 0, 512);
  cvt(eProjW, wEPrj, 512, 512, 512, 0, 512);
  cvt(dProjW, wDPrj, 512, 512, 512, 0, 512);
  cvt(jW1,    wJ1e,  512, 512, 1024, 0,   512);
  cvt(jW1,    wJ1d,  512, 512, 1024, 512, 512);
  cvt(jW2,    wJ2,   1024, 512, 512, 0, 512);

  // Encoder layer 0: xp = xsPad @ Wih0^T + b0 ; scan
  k_gemm<false><<<dim3(128, 16), 128, 0, stream>>>(xsPad, wWih0, eb0, xp, 96, 2048);
  hipMemsetAsync(stateBlk, 0, 49408, stream);
  k_lstm_scan<<<dim3(16), 256, 0, stream>>>(xp, wWhh0, h0seq, hstate, cstate, cnt, gen, 256);

  // Encoder layer 1
  k_gemm<false><<<dim3(128, 16), 128, 0, stream>>>(h0seq, wWih1, eb1, xp, 512, 2048);
  hipMemsetAsync(stateBlk, 0, 49408, stream);
  k_lstm_scan<<<dim3(16), 256, 0, stream>>>(xp, wWhh1, h1seq, hstate, cstate, cnt, gen, 256);

  // Encoder projection (bf16 out)
  k_gemm<true><<<dim3(128, 4), 128, 0, stream>>>(h1seq, wEPrj, eProjb, hEnc, 512, 512);

  // Decoder: embed -> xp -> scan -> projection
  k_embed<<<(528 * 512 + 255) / 256, 256, 0, stream>>>(ys, embed, eDec);
  k_gemm<false><<<dim3(33, 16), 128, 0, stream>>>(eDec, wdWih, db, xp, 512, 2048);
  hipMemsetAsync(stateBlk, 0, 49408, stream);
  hipMemsetAsync(hdseq, 0, (size_t)528 * 512 * 2, stream);  // zero pad rows
  k_lstm_scan<<<dim3(16), 256, 0, stream>>>(xp, wdWhh, hdseq, hstate, cstate, cnt, gen, 65);
  k_gemm<true><<<dim3(33, 4), 128, 0, stream>>>(hdseq, wDPrj, dProjb, hDec, 512, 512);

  // Joint pre-projections (f32, no bias; jb1 folded into tanh in k_joint)
  k_gemm<false><<<dim3(128, 4), 128, 0, stream>>>(hEnc, wJ1e, nullptr, preE, 512, 512);
  k_gemm<false><<<dim3(33, 4), 128, 0, stream>>>(hDec, wJ1d, nullptr, preD, 512, 512);

  // Joint: 8*256*65 = 133120 rows -> 4160 tiles of 32 rows, 1024 cols -> 4 N-blocks
  k_joint<<<dim3(4160, 4), 256, 0, stream>>>(preE, preD, jb1, wJ2, jb2, out);
}
